// PoseMagic_15152644620510
// MI455X (gfx1250) — matrix operand order repaired and checked by reading, not compiled, not run on hardware
//
#include <hip/hip_runtime.h>
#include <hip/hip_bf16.h>
#include <math.h>

#define NL   26
#define D_   128
#define DI_  256
#define DS_  16
#define DTR_ 8
#define DC_  4
#define J_   17
#define DP_  512
#define T_   243
#define B_   2
#define NTOK (B_ * T_ * J_)   // 8262

typedef __attribute__((ext_vector_type(16))) __bf16       v16bf;
typedef __attribute__((ext_vector_type(8)))  float        v8f;
typedef __attribute__((ext_vector_type(8)))  unsigned int v8u;

static __device__ __forceinline__ unsigned short f2bf(float f) {
    unsigned u = __float_as_uint(f);
    unsigned r = u + 0x7FFFu + ((u >> 16) & 1u);   // round-to-nearest-even
    return (unsigned short)(r >> 16);
}
static __device__ __forceinline__ float siluf(float x) { return x / (1.f + expf(-x)); }

// ---------------------------------------------------------------------------
// WMMA bf16 GEMM: C[M,N] = A[M,K] * Bt[N,K]^T (+bias) ; flags: 1=accum, 2=tanh
// One wave computes a 16x32 output tile (two 16x16 WMMA tiles sharing the A
// operand). Out-of-range rows are CLAMPED for loads (no EXEC churn in the hot
// loop; duplicate results are dropped by the guarded store). K % 32 == 0.
// ---------------------------------------------------------------------------
__global__ void k_gemm(const unsigned short* __restrict__ A,
                       const unsigned short* __restrict__ Bt,
                       const float* __restrict__ bias,
                       float* __restrict__ C,
                       int M, int N, int K, int ldc, int flags) {
    const int lane = threadIdx.x;                 // 0..31
    const int tm = blockIdx.y, tn = blockIdx.x;
    int rA = tm * 16 + (lane & 15);               // A row (M)
    if (rA >= M) rA = M - 1;                      // clamp: load-safe, store-guarded
    int n0 = tn * 32 + (lane & 15);
    int n1 = n0 + 16;
    const int rB0 = (n0 < N) ? n0 : (N - 1);
    const int rB1 = (n1 < N) ? n1 : (N - 1);
    const int kh = (lane >> 4) << 3;              // C row half offset: 0 or 8
    const int kb = (lane >> 4) << 4;              // B half offset: 0 or 16

    const unsigned short* pa  = A  + (long)rA  * K + kb;
    const unsigned short* pb0 = Bt + (long)rB0 * K + kb;
    const unsigned short* pb1 = Bt + (long)rB1 * K + kb;

    v8f acc0 = {0.f, 0.f, 0.f, 0.f, 0.f, 0.f, 0.f, 0.f};
    v8f acc1 = {0.f, 0.f, 0.f, 0.f, 0.f, 0.f, 0.f, 0.f};
    for (int k0 = 0; k0 < K; k0 += 32) {
        // A 16-bit layout: lanes0-15 K=[0..15]; lanes16-31 K=[16..31]
        uint4 a0 = *(const uint4*)(pa + k0);
        uint4 a1 = *(const uint4*)(pa + k0 + 8);
        // B 16-bit layout: lanes0-15 K=[0..15]; lanes16-31 K=[16..31]
        uint4 b00 = *(const uint4*)(pb0 + k0);
        uint4 b01 = *(const uint4*)(pb0 + k0 + 8);
        uint4 b10 = *(const uint4*)(pb1 + k0);
        uint4 b11 = *(const uint4*)(pb1 + k0 + 8);
        v8u ra = {a0.x, a0.y, a0.z, a0.w, a1.x, a1.y, a1.z, a1.w};
        v8u r0 = {b00.x, b00.y, b00.z, b00.w, b01.x, b01.y, b01.z, b01.w};
        v8u r1 = {b10.x, b10.y, b10.z, b10.w, b11.x, b11.y, b11.z, b11.w};
        v16bf av  = __builtin_bit_cast(v16bf, ra);
        v16bf bv0 = __builtin_bit_cast(v16bf, r0);
        v16bf bv1 = __builtin_bit_cast(v16bf, r1);
        acc0 = __builtin_amdgcn_wmma_f32_16x16x32_bf16(false, av, false, bv0,
                                                       (short)0, acc0, false, false);
        acc1 = __builtin_amdgcn_wmma_f32_16x16x32_bf16(false, av, false, bv1,
                                                       (short)0, acc1, false, false);
    }
    const int mbase = tm * 16 + kh;               // C row base: +0 or +8
    if (n0 < N) {
        float bvv = bias ? bias[n0] : 0.f;
        #pragma unroll
        for (int r = 0; r < 8; r++) {
            int m = mbase + r;
            if (m < M) {
                float v = acc0[r] + bvv;
                long off = (long)m * ldc + n0;
                if (flags & 1) v += C[off];
                if (flags & 2) v = tanhf(v);
                C[off] = v;
            }
        }
    }
    if (n1 < N) {
        float bvv = bias ? bias[n1] : 0.f;
        #pragma unroll
        for (int r = 0; r < 8; r++) {
            int m = mbase + r;
            if (m < M) {
                float v = acc1[r] + bvv;
                long off = (long)m * ldc + n1;
                if (flags & 1) v += C[off];
                if (flags & 2) v = tanhf(v);
                C[off] = v;
            }
        }
    }
}

// ---------------------------------------------------------------------------
// Weight prep: src f32 [L][K][N] -> dst bf16 [L][N][K] (transposed per layer)
// ---------------------------------------------------------------------------
__global__ void k_convT(const float* __restrict__ src, unsigned short* __restrict__ dst,
                        int Lyr, int K, int N) {
    long idx = (long)blockIdx.x * blockDim.x + threadIdx.x;
    long tot = (long)Lyr * K * N;
    if (idx >= tot) return;
    long l = idx / ((long)K * N);
    long r = idx % ((long)K * N);
    long k = r / N, n = r % N;
    dst[(l * N + n) * K + k] = f2bf(src[idx]);
}

__global__ void k_cvt(const float* __restrict__ src, unsigned short* __restrict__ dst, int n) {
    int idx = blockIdx.x * blockDim.x + threadIdx.x;
    if (idx < n) dst[idx] = f2bf(src[idx]);
}

// ---------------------------------------------------------------------------
// Input embedding: x = x2d @ W_inp + b_inp + pos_embed
// ---------------------------------------------------------------------------
__global__ void k_embed(const float* __restrict__ x2d, const float* __restrict__ W_inp,
                        const float* __restrict__ b_inp, const float* __restrict__ pe,
                        float* __restrict__ x) {
    int idx = blockIdx.x * blockDim.x + threadIdx.x;
    if (idx >= NTOK * D_) return;
    int d = idx & (D_ - 1);
    int tok = idx / D_;
    int j = tok % J_;
    int l = (tok / J_) % T_;
    const float* xi = x2d + (long)tok * 3;
    float v = b_inp[d] + xi[0] * W_inp[d] + xi[1] * W_inp[D_ + d] + xi[2] * W_inp[2 * D_ + d];
    v += pe[((long)l * J_ + j) * D_ + d];
    x[idx] = v;
}

// ---------------------------------------------------------------------------
// Causal depthwise conv (width 4) + SiLU. rev handles the flipped direction.
// Sequence n: base token = (n/div)*os + (n%div)*is ; element p at base+p*stride
// ---------------------------------------------------------------------------
__global__ void k_conv(const float* __restrict__ xz, const float* __restrict__ conv_w,
                       const float* __restrict__ conv_b, float* __restrict__ xc,
                       unsigned short* __restrict__ xc_bf,
                       int Nseq, int L, int div, int os, int is, int stride, int rev) {
    long idx = (long)blockIdx.x * blockDim.x + threadIdx.x;
    long tot = (long)Nseq * L * DI_;
    if (idx >= tot) return;
    int i = idx % DI_;
    long t = idx / DI_;
    int p = t % L;
    int n = t / L;
    long base = (long)(n / div) * os + (long)(n % div) * is;
    float acc = conv_b[i];
    #pragma unroll
    for (int k = 0; k < DC_; k++) {
        int q = p + (rev ? (3 - k) : (k - 3));
        if (q >= 0 && q < L) {
            long tok = base + (long)q * stride;
            acc += conv_w[i * DC_ + k] * xz[tok * (2 * DI_) + i];
        }
    }
    float s = siluf(acc);
    long tokO = base + (long)p * stride;
    xc[tokO * DI_ + i] = s;
    xc_bf[tokO * DI_ + i] = f2bf(s);
}

// dt = softplus(proj[:, :8] @ W_dt + b_dt)
__global__ void k_dt(const float* __restrict__ proj, const float* __restrict__ W_dt,
                     const float* __restrict__ b_dt, float* __restrict__ dt, int ntok) {
    long idx = (long)blockIdx.x * blockDim.x + threadIdx.x;
    if (idx >= (long)ntok * DI_) return;
    int i = idx % DI_;
    long tok = idx / DI_;
    const float* pr = proj + tok * 40;
    float v = b_dt[i];
    #pragma unroll
    for (int r = 0; r < DTR_; r++) v += pr[r] * W_dt[r * DI_ + i];
    dt[idx] = (v > 20.f) ? v : log1pf(expf(v));
}

// ---------------------------------------------------------------------------
// Selective scan. One thread per (sequence, channel), h[16] in registers.
// Writes y * silu(z) as bf16 (input to the W_out WMMA GEMM).
// ---------------------------------------------------------------------------
__global__ void k_scan(const float* __restrict__ xz, const float* __restrict__ xc,
                       const float* __restrict__ proj, const float* __restrict__ dt,
                       const float* __restrict__ A_log, const float* __restrict__ D_skip,
                       unsigned short* __restrict__ yg,
                       int Nseq, int L, int div, int os, int is, int stride, int rev) {
    int idx = blockIdx.x * blockDim.x + threadIdx.x;
    if (idx >= Nseq * DI_) return;
    int i = idx % DI_;
    int n = idx / DI_;
    long base = (long)(n / div) * os + (long)(n % div) * is;
    float a[DS_], h[DS_];
    #pragma unroll
    for (int s = 0; s < DS_; s++) { a[s] = -expf(A_log[i * DS_ + s]); h[s] = 0.f; }
    float dsk = D_skip[i];
    for (int st = 0; st < L; st++) {
        int p = rev ? (L - 1 - st) : st;
        long tok = base + (long)p * stride;
        float dtv = dt[tok * DI_ + i];
        float xv = xc[tok * DI_ + i];
        const float* pr = proj + tok * 40;
        float dtx = dtv * xv;
        float y = dsk * xv;
        #pragma unroll
        for (int s = 0; s < DS_; s++) {
            h[s] = expf(dtv * a[s]) * h[s] + dtx * pr[8 + s];
            y += h[s] * pr[24 + s];
        }
        float zv = xz[tok * (2 * DI_) + DI_ + i];
        yg[tok * DI_ + i] = f2bf(y * siluf(zv));
    }
}

// Row softmax of the 17x17 adjacency
__global__ void k_soft17(const float* __restrict__ adj, float* __restrict__ As) {
    int i = threadIdx.x;
    if (i >= J_) return;
    const float* row = adj + i * J_;
    float mx = row[0];
    for (int j = 1; j < J_; j++) mx = fmaxf(mx, row[j]);
    float e[J_], s = 0.f;
    for (int j = 0; j < J_; j++) { e[j] = expf(row[j] - mx); s += e[j]; }
    for (int j = 0; j < J_; j++) As[i * J_ + j] = e[j] / s;
}

// xg[b,l,i,d] = relu( sum_j As[i,j] * pre[b,l,j,d] )
__global__ void k_gcn(const float* __restrict__ As, const float* __restrict__ pre,
                      float* __restrict__ xg, int ntok) {
    long idx = (long)blockIdx.x * blockDim.x + threadIdx.x;
    if (idx >= (long)ntok * D_) return;
    int d = idx & (D_ - 1);
    long tok = idx / D_;
    int i = tok % J_;
    long bl = tok / J_;
    float s = 0.f;
    for (int j = 0; j < J_; j++) s += As[i * J_ + j] * pre[(bl * J_ + j) * D_ + d];
    xg[idx] = fmaxf(s, 0.f);
}

// Banded temporal averaging (|l-m|<=2, row-normalized) + relu
__global__ void k_tband(const float* __restrict__ pre, float* __restrict__ xtg, int ntok) {
    long idx = (long)blockIdx.x * blockDim.x + threadIdx.x;
    if (idx >= (long)ntok * D_) return;
    int d = idx & (D_ - 1);
    long tok = idx / D_;
    int j = tok % J_;
    long bt = tok / J_;
    int l = bt % T_;
    long b = bt / T_;
    int lo = l - 2 > 0 ? l - 2 : 0;
    int hi = l + 2 < T_ - 1 ? l + 2 : T_ - 1;
    float s = 0.f;
    for (int m = lo; m <= hi; m++) s += pre[(((long)b * T_ + m) * J_ + j) * D_ + d];
    xtg[idx] = fmaxf(s / (float)(hi - lo + 1), 0.f);
}

// 2-way softmax fusion: out = a0*xm + a1*xg  (accum adds into out)
__global__ void k_fusion(const float* __restrict__ xm, const float* __restrict__ xg,
                         const float* __restrict__ Wf, const float* __restrict__ bfv,
                         float* __restrict__ out, int ntok, int accum) {
    int tok = blockIdx.x * blockDim.x + threadIdx.x;
    if (tok >= ntok) return;
    const float* m = xm + (long)tok * D_;
    const float* g = xg + (long)tok * D_;
    float a0 = bfv[0], a1 = bfv[1];
    for (int d = 0; d < D_; d++) { a0 += m[d] * Wf[d * 2];          a1 += m[d] * Wf[d * 2 + 1]; }
    for (int d = 0; d < D_; d++) { a0 += g[d] * Wf[(D_ + d) * 2];   a1 += g[d] * Wf[(D_ + d) * 2 + 1]; }
    float mx = fmaxf(a0, a1);
    float e0 = expf(a0 - mx), e1 = expf(a1 - mx);
    float inv = 1.f / (e0 + e1);
    e0 *= inv; e1 *= inv;
    float* o = out + (long)tok * D_;
    for (int d = 0; d < D_; d++) {
        float v = e0 * m[d] + e1 * g[d];
        if (accum) o[d] += v; else o[d] = v;
    }
}

// Head: out = hidden(tanh, f32 [tok,512]) @ W_head + b_head
__global__ void k_head(const float* __restrict__ hidden, const float* __restrict__ W_head,
                       const float* __restrict__ b_head, float* __restrict__ out, int ntok) {
    long idx = (long)blockIdx.x * blockDim.x + threadIdx.x;
    if (idx >= (long)ntok * 3) return;
    int c = idx % 3;
    long tok = idx / 3;
    const float* h = hidden + tok * DP_;
    float s = b_head[c];
    for (int p = 0; p < DP_; p++) s += h[p] * W_head[p * 3 + c];
    out[idx] = s;
}

// ---------------------------------------------------------------------------
extern "C" void kernel_launch(void* const* d_in, const int* in_sizes, int n_in,
                              void* d_out, int out_size, void* d_ws, size_t ws_size,
                              hipStream_t stream) {
    (void)in_sizes; (void)out_size; (void)ws_size;
    // JAX pytree (sorted-key) flattening of setup_inputs():
    //  0 W_feat  1 W_head  2 W_inp  3 b_feat  4 b_head  5 b_inp
    //  6 Wf  7 Wsg  8 Wtg  9 adj  10 bf  11 bsg  12 btg
    //  mamba dirs (11 leaves each: A_log,D_skip,W_dt,W_in,W_out,W_x,b_dt,b_in,b_out,conv_b,conv_w):
    //  13.. sp_b   24.. sp_f   35.. tp_b   46.. tp_f
    //  57 pos_embed  58 x_2d
    const float* in[59];
    for (int i = 0; i < n_in && i < 59; i++) in[i] = (const float*)d_in[i];

    size_t off = 0;
    auto alloc = [&](size_t elems, size_t esz) -> void* {
        void* p = (char*)d_ws + off;
        off = (off + elems * esz + 255) & ~(size_t)255;
        return p;
    };
    float* xA    = (float*)alloc((size_t)NTOK * D_, 4);
    float* xB    = (float*)alloc((size_t)NTOK * D_, 4);
    float* xz    = (float*)alloc((size_t)NTOK * 2 * DI_, 4);   // also reused as head hidden
    float* xconv = (float*)alloc((size_t)NTOK * DI_, 4);
    float* proj  = (float*)alloc((size_t)NTOK * 40, 4);
    float* dtb   = (float*)alloc((size_t)NTOK * DI_, 4);
    float* ysp   = (float*)alloc((size_t)NTOK * D_, 4);
    float* ytp   = (float*)alloc((size_t)NTOK * D_, 4);
    float* xgpre = (float*)alloc((size_t)NTOK * D_, 4);
    float* xg    = (float*)alloc((size_t)NTOK * D_, 4);
    float* xtg   = (float*)alloc((size_t)NTOK * D_, 4);
    float* Asoft = (float*)alloc(J_ * J_, 4);
    unsigned short* x_bf     = (unsigned short*)alloc((size_t)NTOK * D_, 2);
    unsigned short* xconv_bf = (unsigned short*)alloc((size_t)NTOK * DI_, 2);
    unsigned short* yg_bf    = (unsigned short*)alloc((size_t)NTOK * DI_, 2);
    unsigned short *w_in_t[4], *w_x_t[4], *w_out_t[4];
    for (int d = 0; d < 4; d++) {
        w_in_t[d]  = (unsigned short*)alloc((size_t)NL * 2 * DI_ * D_, 2);  // [l][512][128]
        w_x_t[d]   = (unsigned short*)alloc((size_t)NL * 40 * DI_, 2);     // [l][40][256]
        w_out_t[d] = (unsigned short*)alloc((size_t)NL * D_ * DI_, 2);     // [l][128][256]
    }
    unsigned short* wsg_t   = (unsigned short*)alloc((size_t)NL * D_ * D_, 2);
    unsigned short* wtg_t   = (unsigned short*)alloc((size_t)NL * D_ * D_, 2);
    unsigned short* wfeat_t = (unsigned short*)alloc((size_t)DP_ * D_, 2);

    const int dirBase[4] = {24, 13, 46, 35};   // sp_f, sp_b, tp_f, tp_b

    auto eg = [](long n) { return dim3((unsigned)((n + 255) / 256)); };
    auto gemmGrid = [](int N, int M) { return dim3((unsigned)((N + 31) / 32),
                                                   (unsigned)((M + 15) / 16)); };

    // ---- weight prep (bf16, transposed) ----
    for (int dd = 0; dd < 4; dd++) {
        int pb = dirBase[dd];
        k_convT<<<eg((long)NL * D_ * 2 * DI_), 256, 0, stream>>>(in[pb + 3], w_in_t[dd],  NL, D_,  2 * DI_);
        k_convT<<<eg((long)NL * DI_ * 40),     256, 0, stream>>>(in[pb + 5], w_x_t[dd],   NL, DI_, 40);
        k_convT<<<eg((long)NL * DI_ * D_),     256, 0, stream>>>(in[pb + 4], w_out_t[dd], NL, DI_, D_);
    }
    k_convT<<<eg((long)NL * D_ * D_), 256, 0, stream>>>(in[7], wsg_t, NL, D_, D_);
    k_convT<<<eg((long)NL * D_ * D_), 256, 0, stream>>>(in[8], wtg_t, NL, D_, D_);
    k_convT<<<eg((long)D_ * DP_),     256, 0, stream>>>(in[0], wfeat_t, 1, D_, DP_);

    // ---- input embedding ----
    k_embed<<<eg((long)NTOK * D_), 256, 0, stream>>>(in[58], in[2], in[5], in[57], xA);

    auto mamba = [&](int dd, int l, int rev, int Nseq, int L, int div, int os, int is,
                     int stride, float* ybuf, int accum) {
        int pb = dirBase[dd];
        k_gemm<<<gemmGrid(2 * DI_, NTOK), 32, 0, stream>>>(
            x_bf, w_in_t[dd] + (size_t)l * 2 * DI_ * D_,
            in[pb + 7] + (size_t)l * 2 * DI_, xz, NTOK, 2 * DI_, D_, 2 * DI_, 0);
        k_conv<<<eg((long)Nseq * L * DI_), 256, 0, stream>>>(
            xz, in[pb + 10] + (size_t)l * DI_ * DC_, in[pb + 9] + (size_t)l * DI_,
            xconv, xconv_bf, Nseq, L, div, os, is, stride, rev);
        k_gemm<<<gemmGrid(40, NTOK), 32, 0, stream>>>(
            xconv_bf, w_x_t[dd] + (size_t)l * 40 * DI_, nullptr, proj,
            NTOK, 40, DI_, 40, 0);
        k_dt<<<eg((long)NTOK * DI_), 256, 0, stream>>>(
            proj, in[pb + 2] + (size_t)l * DTR_ * DI_, in[pb + 6] + (size_t)l * DI_, dtb, NTOK);
        k_scan<<<eg((long)Nseq * DI_), 256, 0, stream>>>(
            xz, xconv, proj, dtb, in[pb + 0] + (size_t)l * DI_ * DS_,
            in[pb + 1] + (size_t)l * DI_, yg_bf, Nseq, L, div, os, is, stride, rev);
        k_gemm<<<gemmGrid(D_, NTOK), 32, 0, stream>>>(
            yg_bf, w_out_t[dd] + (size_t)l * D_ * DI_,
            in[pb + 8] + (size_t)l * D_, ybuf, NTOK, D_, DI_, D_, accum);
    };

    float* xcur = xA;
    float* xnxt = xB;

    for (int l = 0; l < NL; l++) {
        k_cvt<<<eg((long)NTOK * D_), 256, 0, stream>>>(xcur, x_bf, NTOK * D_);

        // spatial bimamba (seqs along J: base = n*J, stride 1)
        mamba(0, l, 0, B_ * T_, J_, 1, J_, 0, 1, ysp, 0);
        mamba(1, l, 1, B_ * T_, J_, 1, J_, 0, 1, ysp, 1);
        // spatial GCN
        k_gemm<<<gemmGrid(D_, NTOK), 32, 0, stream>>>(
            x_bf, wsg_t + (size_t)l * D_ * D_, in[11] + (size_t)l * D_,
            xgpre, NTOK, D_, D_, D_, 0);
        k_soft17<<<1, 32, 0, stream>>>(in[9] + (size_t)l * J_ * J_, Asoft);
        k_gcn<<<eg((long)NTOK * D_), 256, 0, stream>>>(Asoft, xgpre, xg, NTOK);
        k_fusion<<<eg(NTOK), 256, 0, stream>>>(ysp, xg, in[6] + (size_t)l * 2 * D_ * 2,
                                               in[10] + (size_t)l * 2, xnxt, NTOK, 0);

        // temporal bimamba (seqs along T: base = (n/J)*T*J + n%J, stride J)
        mamba(2, l, 0, B_ * J_, T_, J_, T_ * J_, 1, J_, ytp, 0);
        mamba(3, l, 1, B_ * J_, T_, J_, T_ * J_, 1, J_, ytp, 1);
        // temporal band conv
        k_gemm<<<gemmGrid(D_, NTOK), 32, 0, stream>>>(
            x_bf, wtg_t + (size_t)l * D_ * D_, in[12] + (size_t)l * D_,
            xgpre, NTOK, D_, D_, D_, 0);
        k_tband<<<eg((long)NTOK * D_), 256, 0, stream>>>(xgpre, xtg, NTOK);
        k_fusion<<<eg(NTOK), 256, 0, stream>>>(ytp, xtg, in[6] + (size_t)l * 2 * D_ * 2,
                                               in[10] + (size_t)l * 2, xnxt, NTOK, 1);

        float* t = xcur; xcur = xnxt; xnxt = t;
    }

    // ---- head ----
    k_cvt<<<eg((long)NTOK * D_), 256, 0, stream>>>(xcur, x_bf, NTOK * D_);
    k_gemm<<<gemmGrid(DP_, NTOK), 32, 0, stream>>>(
        x_bf, wfeat_t, in[3], xz, NTOK, DP_, D_, DP_, 2); // tanh
    k_head<<<eg((long)NTOK * 3), 256, 0, stream>>>(xz, in[1], in[4], (float*)d_out, NTOK);
}
